// HungarianMatcher_20203526160786
// MI455X (gfx1250) — matrix operand order repaired and checked by reading, not compiled, not run on hardware
//
#include <hip/hip_runtime.h>
#include <hip/hip_bf16.h>

// ---------------------------------------------------------------------------
// HungarianMatcher for MI455X (gfx1250, wave32, WMMA)
//
// Shapes (from reference setup_inputs): B=32, Q=300, NCLS=92, M=50
//   inputs : pred_logits (B,Q,92) f32 | pred_boxes (B,Q,4) f32
//            tgt_labels (B,M) i32    | tgt_boxes (B,M,4) f32
//   outputs: pred_idx (B,M) , gt_idx (B,M) , C (B,Q,B*M)  -- flat in d_out
//
// Strategy:
//  - class cost  cc[i,j] = -probs[i, tl[j]]  ==  -(probs @ onehot(tl))[i,j]
//    -> computed with v_wmma_f32_16x16x32_f16 (K padded 92->96, 3 k-steps),
//       one-hot B fragment synthesized in registers (zero memory traffic).
//  - L1 + GIoU terms fused in the WMMA epilogue; C streamed out once
//    (61.4 MB -> ~2.6 us at 23.3 TB/s is the true floor for this op).
//  - Hungarian: 32 independent 50x300 augmenting-path solves, one wave32
//    per batch, lane-parallel column scans + shuffle argmin.
// ---------------------------------------------------------------------------

typedef _Float16 v16h __attribute__((ext_vector_type(16)));
typedef _Float16 v8h  __attribute__((ext_vector_type(8)));
typedef float    v8f  __attribute__((ext_vector_type(8)));

#define B_   32
#define Q_   300
#define NCLS 92
#define M_   50
#define BQ   (B_ * Q_)    // 9600
#define BM   (B_ * M_)    // 1600
#define KP   96           // NCLS padded to multiple of 32
#define HINF 1e30f

// ---------------------------------------------------------------------------
// Kernel 1: softmax over 92 classes, one wave32 per row, emit f16 padded K=96
// ---------------------------------------------------------------------------
__global__ __launch_bounds__(256) void hm_softmax_kernel(
    const float* __restrict__ logits, _Float16* __restrict__ probs)
{
    const int wid  = (blockIdx.x * (blockDim.x >> 5)) + (threadIdx.x >> 5);
    const int lane = threadIdx.x & 31;
    if (wid >= BQ) return;

    const float* row = logits + (size_t)wid * NCLS;
    float x0 = row[lane];
    float x1 = row[lane + 32];
    float x2 = (lane < NCLS - 64) ? row[lane + 64] : -HINF;

    float mx = fmaxf(fmaxf(x0, x1), x2);
    #pragma unroll
    for (int off = 16; off > 0; off >>= 1)
        mx = fmaxf(mx, __shfl_xor(mx, off, 32));

    float e0 = __expf(x0 - mx);
    float e1 = __expf(x1 - mx);
    float e2 = (lane < NCLS - 64) ? __expf(x2 - mx) : 0.0f;

    float s = e0 + e1 + e2;
    #pragma unroll
    for (int off = 16; off > 0; off >>= 1)
        s += __shfl_xor(s, off, 32);

    const float inv = 1.0f / s;
    _Float16* prow = probs + (size_t)wid * KP;
    prow[lane]      = (_Float16)(e0 * inv);
    prow[lane + 32] = (_Float16)(e1 * inv);
    // lanes 28..31 land on columns 92..95: zero pad (e2 already 0 there)
    prow[lane + 64] = (_Float16)(e2 * inv);
}

// ---------------------------------------------------------------------------
// Kernel 2: fused cost matrix. One wave32 per 16x16 tile of C (600x100 tiles).
//   GEMM part: acc = probs(16x96 f16) x onehot(96x16 f16) via 3 WMMA steps.
//   Epilogue : C = -acc + 5*L1 - 2*GIoU, written once, coalesced.
// ---------------------------------------------------------------------------
__global__ __launch_bounds__(128) void hm_cost_wmma_kernel(
    const _Float16* __restrict__ probs,      // BQ x KP
    const float*    __restrict__ pred_boxes, // BQ x 4 (cxcywh)
    const int*      __restrict__ tgt_labels, // BM
    const float*    __restrict__ tgt_boxes,  // BM x 4 (cxcywh)
    float*          __restrict__ Cout)       // BQ x BM
{
    const int TI = BQ / 16;  // 600
    const int TJ = BM / 16;  // 100
    const int wib  = threadIdx.x >> 5;
    const int lane = threadIdx.x & 31;
    const int tile = blockIdx.x * 4 + wib;
    if (tile >= TI * TJ) return;
    const int ti = tile / TJ;
    const int tj = tile - ti * TJ;

    const int nloc = lane & 15;   // N index within tile (and A-row for loads)
    const int kh   = lane >> 4;   // which K-half this lane holds

    // ---- A fragment source: row of probs (ISA 16-bit A 16x32 layout:
    //      lane m<16 holds K 0..7 & 16..23; lane m+16 holds K 8..15 & 24..31)
    const _Float16* arow = probs + (size_t)(ti * 16 + nloc) * KP;
    // ---- B fragment: one-hot of target label for column tj*16+nloc
    //      (same K order as A: lanes 0-15 hold K 0..7 & 16..23, lanes 16-31 hold K 8..15 & 24..31)
    const int jcol = tj * 16 + nloc;
    const int tlab = tgt_labels[jcol];

    v8f acc = {};
    #pragma unroll
    for (int ks = 0; ks < 3; ++ks) {
        const int kb = ks * 32;
        const v8h a0 = *(const v8h*)(arow + kb + kh * 8);
        const v8h a1 = *(const v8h*)(arow + kb + 16 + kh * 8);
        v16h afrag;
        #pragma unroll
        for (int e = 0; e < 8; ++e) { afrag[e] = a0[e]; afrag[8 + e] = a1[e]; }

        v16h bfrag;
        #pragma unroll
        for (int e = 0; e < 16; ++e) {
            const int k = kb + kh * 8 + e + ((e >= 8) ? 8 : 0);
            bfrag[e] = (tlab == k) ? (_Float16)1.0f : (_Float16)0.0f;
        }
        acc = __builtin_amdgcn_wmma_f32_16x16x32_f16(
            false, afrag, false, bfrag, (short)0, acc, false, false);
    }

    // ---- Epilogue. C/D layout: VGPR r, lanes 0-15 -> M=r, lanes 16-31 -> M=r+8
    const float tcx = tgt_boxes[jcol * 4 + 0];
    const float tcy = tgt_boxes[jcol * 4 + 1];
    const float tw  = tgt_boxes[jcol * 4 + 2];
    const float th  = tgt_boxes[jcol * 4 + 3];
    const float bx0 = tcx - 0.5f * tw, by0 = tcy - 0.5f * th;
    const float bx1 = tcx + 0.5f * tw, by1 = tcy + 0.5f * th;
    const float area_b = (bx1 - bx0) * (by1 - by0);

    #pragma unroll
    for (int r = 0; r < 8; ++r) {
        const int i = ti * 16 + r + kh * 8;
        const float pcx = pred_boxes[i * 4 + 0];
        const float pcy = pred_boxes[i * 4 + 1];
        const float pw  = pred_boxes[i * 4 + 2];
        const float ph  = pred_boxes[i * 4 + 3];

        const float cb = fabsf(pcx - tcx) + fabsf(pcy - tcy) +
                         fabsf(pw - tw) + fabsf(ph - th);

        const float ax0 = pcx - 0.5f * pw, ay0 = pcy - 0.5f * ph;
        const float ax1 = pcx + 0.5f * pw, ay1 = pcy + 0.5f * ph;
        const float area_a = (ax1 - ax0) * (ay1 - ay0);

        const float iw = fmaxf(fminf(ax1, bx1) - fmaxf(ax0, bx0), 0.0f);
        const float ih = fmaxf(fminf(ay1, by1) - fmaxf(ay0, by0), 0.0f);
        const float inter = iw * ih;
        const float uni   = area_a + area_b - inter;
        const float iou   = inter / uni;

        const float ew = fmaxf(fmaxf(ax1, bx1) - fminf(ax0, bx0), 0.0f);
        const float eh = fmaxf(fmaxf(ay1, by1) - fminf(ay0, by0), 0.0f);
        const float ae = ew * eh;
        const float giou = iou - (ae - uni) / ae;

        // C = COST_CLASS*(-p) + COST_BBOX*cb + COST_GIOU*(-giou)
        Cout[(size_t)i * BM + jcol] = -acc[r] + 5.0f * cb - 2.0f * giou;
    }
}

// ---------------------------------------------------------------------------
// Kernel 3: Hungarian (augmenting path), one wave32 per batch.
//   cost(i,j) = C[(b*Q + j)*BM + b*M + i],  n=50 rows, m=300 cols
// ---------------------------------------------------------------------------
__global__ __launch_bounds__(32) void hm_hungarian_kernel(
    const float* __restrict__ C, float* __restrict__ pred_out,
    float* __restrict__ gt_out)
{
    const int b    = blockIdx.x;
    const int lane = threadIdx.x;
    const int n = M_, m = Q_;

    __shared__ float u[M_ + 1];
    __shared__ float v[Q_ + 1];
    __shared__ float minv[Q_ + 1];
    __shared__ int   p[Q_ + 1];
    __shared__ int   way[Q_ + 1];
    __shared__ int   used[Q_ + 1];
    __shared__ int   qcol[M_];

    for (int j = lane; j <= m; j += 32) { v[j] = 0.0f; p[j] = -1; way[j] = m; }
    for (int i = lane; i <= n; i += 32) u[i] = 0.0f;
    for (int i = lane; i <  n; i += 32) qcol[i] = 0;
    __syncthreads();

    const float* Cb = C + (size_t)b * Q_ * BM + (size_t)b * M_;
    // cost(i, j) = Cb[j*BM + i]

    for (int i = 0; i < n; ++i) {
        if (lane == 0) p[m] = i;
        for (int j = lane; j <= m; j += 32) { minv[j] = HINF; used[j] = 0; }
        __syncthreads();

        int j0 = m;
        while (p[j0] != -1) {
            if (lane == 0) used[j0] = 1;
            __syncthreads();
            const int   i0  = p[j0];
            const float ui0 = u[i0];

            float bestv = HINF;
            int   bestj = 0;
            for (int j = lane; j < m; j += 32) {
                if (!used[j]) {
                    const float cur = Cb[(size_t)j * BM + i0] - ui0 - v[j];
                    if (cur < minv[j]) { minv[j] = cur; way[j] = j0; }
                    const float mv = minv[j];
                    if (mv < bestv) { bestv = mv; bestj = j; }
                }
            }
            #pragma unroll
            for (int off = 16; off > 0; off >>= 1) {
                const float ov = __shfl_down(bestv, off, 32);
                const int   oj = __shfl_down(bestj, off, 32);
                if (ov < bestv) { bestv = ov; bestj = oj; }
            }
            bestv = __shfl(bestv, 0, 32);
            bestj = __shfl(bestj, 0, 32);
            const float delta = bestv;
            __syncthreads();

            for (int j = lane; j <= m; j += 32) {
                if (used[j]) {
                    u[p[j]] += delta;   // matched rows are distinct -> race-free
                    v[j]    -= delta;
                } else if (j < m) {
                    minv[j] -= delta;
                }
            }
            __syncthreads();
            j0 = bestj;
        }
        if (lane == 0) {
            while (j0 != m) {
                const int j1 = way[j0];
                p[j0] = p[j1];
                j0 = j1;
            }
        }
        __syncthreads();
    }

    // col_of_row: invert the matching
    for (int j = lane; j < m; j += 32) {
        const int r = p[j];
        if (r >= 0 && r < n) qcol[r] = j;
    }
    __syncthreads();

    // argsort(q): columns are distinct, so rank = #{q[k] < q[i]}
    for (int i = lane; i < n; i += 32) {
        const int qi = qcol[i];
        int rank = 0;
        for (int k = 0; k < n; ++k) rank += (qcol[k] < qi) ? 1 : 0;
        pred_out[b * n + rank] = (float)qi;
        gt_out[b * n + rank]   = (float)i;
    }
}

// ---------------------------------------------------------------------------
extern "C" void kernel_launch(void* const* d_in, const int* in_sizes, int n_in,
                              void* d_out, int out_size, void* d_ws,
                              size_t ws_size, hipStream_t stream)
{
    const float* pred_logits = (const float*)d_in[0];  // B*Q*NCLS
    const float* pred_boxes  = (const float*)d_in[1];  // B*Q*4
    const int*   tgt_labels  = (const int*)d_in[2];    // B*M
    const float* tgt_boxes   = (const float*)d_in[3];  // B*M*4

    float* out      = (float*)d_out;
    float* pred_idx = out;                 // B*M
    float* gt_idx   = out + B_ * M_;       // B*M
    float* Cmat     = out + 2 * B_ * M_;   // BQ*BM

    _Float16* probs = (_Float16*)d_ws;     // BQ*KP f16 (1.84 MB)

    // 9600 rows, 8 waves/block
    hm_softmax_kernel<<<BQ / 8, 256, 0, stream>>>(pred_logits, probs);

    // 600*100 = 60000 tiles, 4 waves/block
    hm_cost_wmma_kernel<<<(600 * 100) / 4, 128, 0, stream>>>(
        probs, pred_boxes, tgt_labels, tgt_boxes, Cmat);

    // one wave per batch
    hm_hungarian_kernel<<<B_, 32, 0, stream>>>(Cmat, pred_idx, gt_idx);
}
